// DeepTactile_69148973465953
// MI455X (gfx1250) — matrix operand order repaired and checked by reading, not compiled, not run on hardware
//
#include <hip/hip_runtime.h>
#include <hip/hip_bf16.h>

typedef _Float16 half16  __attribute__((ext_vector_type(16)));
typedef _Float16 half8   __attribute__((ext_vector_type(8)));
typedef float    float8  __attribute__((ext_vector_type(8)));
typedef float    float4v __attribute__((ext_vector_type(4)));

#define B_ 16
#define N_ 100
#define T_ 50
#define E_ 800
#define ROWS_BNT (B_ * N_ * T_)   /* 80000 */
#define MT_BNT   (ROWS_BNT / 16)  /* 5000  */
#define NFEAT    (N_ * 176)       /* 17600 */

__device__ __forceinline__ float8 wmma_f16(half16 a, half16 b, float8 c) {
  // v_wmma_f32_16x16x32_f16 : D = A(16x32) * B(32x16) + C, f32 accumulate
  return __builtin_amdgcn_wmma_f32_16x16x32_f16(false, a, false, b, (short)0, c,
                                                false, false);
}

__device__ __forceinline__ half16 join8(half8 lo, half8 hi) {
  return __builtin_shufflevector(lo, hi, 0, 1, 2, 3, 4, 5, 6, 7, 8, 9, 10, 11,
                                 12, 13, 14, 15);
}

// ---------------------------------------------------------------------------
// Graph prep: degree, enorm = dinv[row]*dinv[col], deterministic CSR by col.
// ---------------------------------------------------------------------------
__global__ void k_graph(const int* __restrict__ ei, float* __restrict__ enorm,
                        int* __restrict__ csr_off, int* __restrict__ csr_eid) {
  const int* row = ei;
  const int* col = ei + E_;
  __shared__ int   deg[N_];
  __shared__ int   soff[N_ + 1];
  __shared__ int   cur[N_];
  __shared__ float dinv[N_];
  int tid = threadIdx.x;
  if (tid < N_) deg[tid] = 0;
  __syncthreads();
  for (int e = tid; e < E_; e += blockDim.x) atomicAdd(&deg[col[e]], 1);
  __syncthreads();
  if (tid < N_) dinv[tid] = (deg[tid] > 0) ? rsqrtf((float)deg[tid]) : 0.f;
  if (tid == 0) {
    int run = 0;
    for (int n = 0; n < N_; ++n) { soff[n] = run; cur[n] = run; run += deg[n]; }
    soff[N_] = run;
  }
  __syncthreads();
  for (int e = tid; e < E_; e += blockDim.x)
    enorm[e] = dinv[row[e]] * dinv[col[e]];
  if (tid <= N_) csr_off[tid] = soff[tid];
  if (tid == 0) {  // serial scatter: deterministic edge order within buckets
    for (int e = 0; e < E_; ++e) { int c = col[e]; csr_eid[cur[c]++] = e; }
  }
}

// ---------------------------------------------------------------------------
// Transpose harness input [B,N,C=2,T] -> channels-last [B,N,T,2].
// ---------------------------------------------------------------------------
__global__ void k_tr_data(const float* __restrict__ Din, float* __restrict__ O) {
  int idx = blockIdx.x * blockDim.x + threadIdx.x;
  int total = B_ * N_ * T_ * 2;
  if (idx >= total) return;
  int c = idx & 1;
  int rest = idx >> 1;
  int t = rest % T_;
  int bn = rest / T_;
  O[idx] = Din[((size_t)bn * 2 + c) * T_ + t];
}

// ---------------------------------------------------------------------------
// Weight prep: f32 [cin, cout] -> transposed f16 [cout, cinp] zero-padded.
// Padding makes the GEMM K-loop tail-free and check-free.
// ---------------------------------------------------------------------------
__global__ void k_prepw(const float* __restrict__ W, _Float16* __restrict__ Wt,
                        int cin, int cinp, int cout) {
  int idx = blockIdx.x * blockDim.x + threadIdx.x;
  int total = cout * cinp;
  if (idx >= total) return;
  int o = idx / cinp;
  int k = idx - o * cinp;
  Wt[idx] = (k < cin) ? (_Float16)W[(size_t)k * cout + o] : (_Float16)0.f;
}

// ---------------------------------------------------------------------------
// Graph propagation (channels-last): Y[b,n,t,c] = sum enorm * X[b,row,t,c].
// c fastest across threads -> fully coalesced.
// ---------------------------------------------------------------------------
__global__ void k_prop(const float* __restrict__ X, int cit, int cio, int C,
                       float* __restrict__ Y, const int* __restrict__ rowidx,
                       const float* __restrict__ enorm,
                       const int* __restrict__ csr_off,
                       const int* __restrict__ csr_eid) {
  int idx = blockIdx.x * blockDim.x + threadIdx.x;
  int total = B_ * N_ * T_ * C;
  if (idx >= total) return;
  int c = idx % C;
  int rest = idx / C;
  int t = rest % T_;
  int r2 = rest / T_;
  int n = r2 % N_;
  int b = r2 / N_;
  float acc = 0.f;
  int e0 = csr_off[n], e1 = csr_off[n + 1];
  for (int j = e0; j < e1; ++j) {
    int e = csr_eid[j];
    int src = rowidx[e];
    acc += enorm[e] * X[((size_t)(b * N_ + src) * T_ + t) * cit + cio + c];
  }
  Y[((size_t)(b * N_ + n) * T_ + t) * C + c] = acc;
}

// ---------------------------------------------------------------------------
// ebnorm statistics (channels-last): per-channel mean/var over all rows.
// ---------------------------------------------------------------------------
__global__ void k_stats(const float* __restrict__ X, int cit, int cio,
                        const float* __restrict__ g, const float* __restrict__ bta,
                        float* __restrict__ stats) {
  int c = blockIdx.x, tid = threadIdx.x;
  float s = 0.f, s2 = 0.f;
  for (int i = tid; i < ROWS_BNT; i += 256) {
    float v = X[(size_t)i * cit + cio + c];
    s += v; s2 += v * v;
  }
  __shared__ float rs[256], rq[256];
  rs[tid] = s; rq[tid] = s2;
  __syncthreads();
  for (int off = 128; off; off >>= 1) {
    if (tid < off) { rs[tid] += rs[tid + off]; rq[tid] += rq[tid + off]; }
    __syncthreads();
  }
  if (tid == 0) {
    const float inv = 1.f / (float)ROWS_BNT;
    float mu  = rs[0] * inv;
    float var = rq[0] * inv - mu * mu;
    float sc  = g[c] * rsqrtf(var + 1e-5f);
    stats[2 * c]     = sc;
    stats[2 * c + 1] = bta[c] - mu * sc;
  }
}

// ---------------------------------------------------------------------------
// Fused affine-norm + LIF scan over time (channels-last; coalesced across c).
// ---------------------------------------------------------------------------
__global__ void k_normlif(const float* __restrict__ X, int cit, int cio, int C,
                          float* __restrict__ Y, int cot, int coo,
                          const float* __restrict__ stats) {
  int idx = blockIdx.x * blockDim.x + threadIdx.x;
  int total = B_ * N_ * C;
  if (idx >= total) return;
  int c = idx % C, bn = idx / C;
  float sc = stats[2 * c], sh = stats[2 * c + 1];
  const float* xin = X + (size_t)bn * T_ * cit + cio + c;
  float*       yo  = Y + (size_t)bn * T_ * cot + coo + c;
  float mem = 0.f, sp = 0.f;
  for (int t = 0; t < T_; ++t) {
    float v = xin[(size_t)t * cit] * sc + sh;
    mem = mem * 0.2f * (1.f - sp) + v;
    sp  = (mem > 0.5f) ? 1.f : 0.f;
    yo[(size_t)t * cot] = sp;
  }
}

// ---------------------------------------------------------------------------
// Final norm+LIF writing f16 spikes directly in fc1 GEMM layout [t*16+b, f],
// f = n*176 + c. Feeds k_gemm_tb with zero-conversion contiguous A rows.
// ---------------------------------------------------------------------------
__global__ void k_normlif_fc(const float* __restrict__ X,
                             const float* __restrict__ stats,
                             _Float16* __restrict__ D) {
  int idx = blockIdx.x * blockDim.x + threadIdx.x;
  int total = B_ * N_ * 176;
  if (idx >= total) return;
  int c = idx % 176;
  int n = (idx / 176) % N_;
  int b = idx / (176 * N_);
  float sc = stats[2 * c], sh = stats[2 * c + 1];
  const float* xin = X + (size_t)(b * N_ + n) * T_ * 176 + c;
  _Float16* dout = D + (size_t)b * NFEAT + (size_t)n * 176 + c;
  float mem = 0.f, sp = 0.f;
  for (int t = 0; t < T_; ++t) {
    float v = xin[(size_t)t * 176] * sc + sh;
    mem = mem * 0.2f * (1.f - sp) + v;
    sp  = (mem > 0.5f) ? 1.f : 0.f;
    dout[(size_t)t * 16 * NFEAT] = (_Float16)sp;
  }
}

// ---------------------------------------------------------------------------
// WMMA GEMM over M=80000 channels-last rows. A fragment = 4 x b128 vector
// loads (AVEC) or clamped scalar path (conv0, cit=2). B fragment = 2 x 16B
// f16 vector loads from pre-transposed zero-padded weights. No tail, no
// bounds checks in the hot loop. NTW output tiles share one A fragment.
// ---------------------------------------------------------------------------
template <int COUT, int NTW, bool ACCUM, bool AVEC>
__global__ void k_gemm_bnt(const float* __restrict__ X, int cit, int cio,
                           int cin, int cinp,
                           const _Float16* __restrict__ Wt,
                           const float* __restrict__ bias,
                           float* __restrict__ Y, int cot, int coo) {
  constexpr int TN  = COUT / 16;
  constexpr int GPM = TN / NTW;  // wave-groups per m-tile
  int gtid = blockIdx.x * blockDim.x + threadIdx.x;
  int wave = gtid >> 5;
  int lane = threadIdx.x & 31;
  if (wave >= MT_BNT * GPM) return;
  int mtile = wave / GPM;
  int ng    = wave - mtile * GPM;
  int l = lane & 15, hi = lane >> 4;
  const float* Ab = X + (size_t)(mtile * 16 + l) * cit + cio;
  int col0 = ng * (NTW * 16) + l;

  float8 acc[NTW];
#pragma unroll
  for (int n = 0; n < NTW; ++n)
    acc[n] = (float8){0.f, 0.f, 0.f, 0.f, 0.f, 0.f, 0.f, 0.f};

  for (int kk = 0; kk < cinp; kk += 32) {
    half16 a;
    if (AVEC) {
      const float* pa = Ab + kk + hi * 8;
      float4v a0 = *(const float4v*)(pa);
      float4v a1 = *(const float4v*)(pa + 4);
      float4v a2 = *(const float4v*)(pa + 16);
      float4v a3 = *(const float4v*)(pa + 20);
#pragma unroll
      for (int j = 0; j < 4; ++j) {
        a[j]      = (_Float16)a0[j];
        a[j + 4]  = (_Float16)a1[j];
        a[j + 8]  = (_Float16)a2[j];
        a[j + 12] = (_Float16)a3[j];
      }
    } else {  // conv0 path: clamp addresses, zero-select (weights padded too)
      int ka = kk + hi * 8;
#pragma unroll
      for (int j = 0; j < 8; ++j) {
        int k0 = ka + j, k1 = ka + 16 + j;
        float v0 = Ab[k0 < cin ? k0 : cin - 1];
        float v1 = Ab[k1 < cin ? k1 : cin - 1];
        a[j]     = (_Float16)(k0 < cin ? v0 : 0.f);
        a[j + 8] = (_Float16)(k1 < cin ? v1 : 0.f);
      }
    }
#pragma unroll
    for (int n = 0; n < NTW; ++n) {
      const _Float16* pw = Wt + (size_t)(col0 + n * 16) * cinp + kk + hi * 8;
      half8 b0 = *(const half8*)(pw);
      half8 b1 = *(const half8*)(pw + 16);
      acc[n] = wmma_f16(a, join8(b0, b1), acc[n]);
    }
  }

  float bvs[NTW];
#pragma unroll
  for (int n = 0; n < NTW; ++n) bvs[n] = bias ? bias[col0 + n * 16] : 0.f;
#pragma unroll
  for (int v = 0; v < 8; ++v) {
    int row = mtile * 16 + v + (hi << 3);
    float* base = Y + (size_t)row * cot + coo;
#pragma unroll
    for (int n = 0; n < NTW; ++n) {
      float* p = base + col0 + n * 16;
      float val = acc[n][v] + bvs[n];
      if (ACCUM) val += *p;
      *p = val;
    }
  }
}

// ---------------------------------------------------------------------------
// FC1 pre-GEMM hoisted out of the temporal scan: M=800 rows (t,b), K=17600,
// N=128. A is f16 spikes [800,17600]; W is pre-transposed f16 [128,17600].
// All fragment loads are contiguous 16B vector loads; no conversions.
// ---------------------------------------------------------------------------
__global__ void k_gemm_tb(const _Float16* __restrict__ S,
                          const _Float16* __restrict__ Wt,
                          const float* __restrict__ bias, float* __restrict__ Z) {
  int gtid = blockIdx.x * blockDim.x + threadIdx.x;
  int wave = gtid >> 5;
  int lane = threadIdx.x & 31;
  if (wave >= T_ * 2) return;  // 50 m-tiles x 2 groups of 4 n-tiles
  int mtile = wave >> 1;
  int ng    = wave & 1;
  int l = lane & 15, hi = lane >> 4;
  const _Float16* Ab = S + (size_t)(mtile * 16 + l) * NFEAT;
  int col0 = ng * 64 + l;
  float8 acc[4];
#pragma unroll
  for (int n = 0; n < 4; ++n)
    acc[n] = (float8){0.f, 0.f, 0.f, 0.f, 0.f, 0.f, 0.f, 0.f};
  for (int kk = 0; kk < NFEAT; kk += 32) {
    const _Float16* pa = Ab + kk + hi * 8;
    half8 x0 = *(const half8*)(pa);
    half8 x1 = *(const half8*)(pa + 16);
    half16 a = join8(x0, x1);
#pragma unroll
    for (int n = 0; n < 4; ++n) {
      const _Float16* pw = Wt + (size_t)(col0 + n * 16) * NFEAT + kk + hi * 8;
      half8 b0 = *(const half8*)(pw);
      half8 b1 = *(const half8*)(pw + 16);
      acc[n] = wmma_f16(a, join8(b0, b1), acc[n]);
    }
  }
  float bvs[4];
#pragma unroll
  for (int n = 0; n < 4; ++n) bvs[n] = bias[col0 + n * 16];
#pragma unroll
  for (int v = 0; v < 8; ++v) {
    int row = mtile * 16 + v + (hi << 3);
#pragma unroll
    for (int n = 0; n < 4; ++n)
      Z[(size_t)row * 128 + col0 + n * 16] = acc[n][v] + bvs[n];
  }
}

// ---------------------------------------------------------------------------
// Zero-pad fc3 weights [256,36] -> [256,48] and bias [36] -> [48].
// ---------------------------------------------------------------------------
__global__ void k_pad_fc3(const float* __restrict__ W3, const float* __restrict__ B3,
                          float* __restrict__ W3p, float* __restrict__ B3p) {
  int i = blockIdx.x * blockDim.x + threadIdx.x;
  if (i < 256 * 48) {
    int k = i / 48, o = i - k * 48;
    W3p[i] = (o < 36) ? W3[k * 36 + o] : 0.f;
  }
  if (i < 48) B3p[i] = (i < 36) ? B3[i] : 0.f;
}

// ---------------------------------------------------------------------------
// Temporal FC recurrence: single persistent workgroup (8 wave32s).
// ---------------------------------------------------------------------------
__global__ void k_fc_scan(const float* __restrict__ Z1,
                          const float* __restrict__ W2, const float* __restrict__ B2,
                          const float* __restrict__ W3p, const float* __restrict__ B3p,
                          float* __restrict__ out) {
  __shared__ float s1[16 * 128], m1[16 * 128];
  __shared__ float s2[16 * 256], m2[16 * 256];
  __shared__ float m3[16 * 48], s3[16 * 48], accb[16 * 48];
  int tid = threadIdx.x;
  int lane = tid & 31, wave = tid >> 5;
  int l = lane & 15, hi = lane >> 4;
  for (int i = tid; i < 16 * 128; i += 256) { s1[i] = 0.f; m1[i] = 0.f; }
  for (int i = tid; i < 16 * 256; i += 256) { s2[i] = 0.f; m2[i] = 0.f; }
  for (int i = tid; i < 16 * 48; i += 256) { s3[i] = 0.f; m3[i] = 0.f; accb[i] = 0.f; }
  __syncthreads();
  for (int t = 0; t < T_; ++t) {
    for (int i = tid; i < 16 * 128; i += 256) {
      float m = m1[i] * 0.2f * (1.f - s1[i]) +
                Z1[((size_t)t * 16 + (i >> 7)) * 128 + (i & 127)];
      m1[i] = m;
      s1[i] = (m > 0.5f) ? 1.f : 0.f;
    }
    __syncthreads();
    for (int nt = wave; nt < 16; nt += 8) {
      int col = nt * 16 + l;
      float8 acc = {0.f, 0.f, 0.f, 0.f, 0.f, 0.f, 0.f, 0.f};
      for (int kk = 0; kk < 128; kk += 32) {
        const float* pa = &s1[l * 128 + kk + hi * 8];
        half16 a, b;
#pragma unroll
        for (int j = 0; j < 8; ++j) {
          a[j]     = (_Float16)pa[j];
          a[j + 8] = (_Float16)pa[16 + j];
        }
        const float* pw = W2 + (size_t)(kk + hi * 8) * 256 + col;
#pragma unroll
        for (int j = 0; j < 16; ++j) b[j] = (_Float16)pw[(j < 8 ? j : j + 8) * 256];
        acc = wmma_f16(a, b, acc);
      }
      float bv = B2[col];
#pragma unroll
      for (int v = 0; v < 8; ++v) {
        int i = (v + (hi << 3)) * 256 + col;  // unique element per lane
        float m = m2[i] * 0.2f * (1.f - s2[i]) + acc[v] + bv;
        m2[i] = m;
        s2[i] = (m > 0.5f) ? 1.f : 0.f;
      }
    }
    __syncthreads();
    if (wave < 3) {
      int col = wave * 16 + l;
      float8 acc = {0.f, 0.f, 0.f, 0.f, 0.f, 0.f, 0.f, 0.f};
      for (int kk = 0; kk < 256; kk += 32) {
        const float* pa = &s2[l * 256 + kk + hi * 8];
        half16 a, b;
#pragma unroll
        for (int j = 0; j < 8; ++j) {
          a[j]     = (_Float16)pa[j];
          a[j + 8] = (_Float16)pa[16 + j];
        }
        const float* pw = W3p + (size_t)(kk + hi * 8) * 48 + col;
#pragma unroll
        for (int j = 0; j < 16; ++j) b[j] = (_Float16)pw[(j < 8 ? j : j + 8) * 48];
        acc = wmma_f16(a, b, acc);
      }
      float bv = B3p[col];
#pragma unroll
      for (int v = 0; v < 8; ++v) {
        int i = (v + (hi << 3)) * 48 + col;
        float m = m3[i] * 0.2f * (1.f - s3[i]) + acc[v] + bv;
        m3[i] = m;
        float sp = (m > 0.5f) ? 1.f : 0.f;
        s3[i] = sp;
        accb[i] += sp;
      }
    }
    __syncthreads();
  }
  for (int i = tid; i < 16 * 36; i += 256) {
    int b = i / 36, o = i - b * 36;
    out[i] = accb[b * 48 + o] * (1.f / (float)T_);
  }
}

// ---------------------------------------------------------------------------
// Host orchestration
// ---------------------------------------------------------------------------
extern "C" void kernel_launch(void* const* d_in, const int* in_sizes, int n_in,
                              void* d_out, int out_size, void* d_ws, size_t ws_size,
                              hipStream_t stream) {
  // params flattened in jax tree order (dict keys sorted):
  //  0: data   1: edge_index
  //  2..73: blocks[bi][li] x 12 leaves:
  //     +0 conv1.b +1 conv1.ws0 +2 conv1.ws1 +3 conv2.b +4..7 conv2.ws0..3
  //     +8 norm1.b +9 norm1.g +10 norm2.b +11 norm2.g
  //  74: conv0.b 75: conv0.ws0 76: conv0.ws1
  //  77: fc1.b 78: fc1.w 79: fc2.b 80: fc2.w 81: fc3.b 82: fc3.w
  //  83: norm0.b 84: norm0.g 85: normL.b 86: normL.g
  //  87: tr.conv.b 88: tr.conv.ws0 89: tr.conv.ws1 90: tr.norm.b 91: tr.norm.g
  (void)in_sizes; (void)n_in; (void)out_size;
  const float* data = (const float*)d_in[0];
  const int*   ei   = (const int*)d_in[1];
  auto P = [&](int i) { return (const float*)d_in[i]; };

  // workspace carve-out (float units; +64 slack on GEMM inputs for padded K)
  float* f = (float*)d_ws;
  size_t off = 0;
  auto carve = [&](size_t n) { float* p = f + off; off += n; return p; };
  float* BUF0  = carve((size_t)ROWS_BNT * 160);        // block0 concat (ch-last)
  float* BUF1  = carve((size_t)ROWS_BNT * 176);        // block1 concat (ch-last)
  float* T0b   = carve((size_t)ROWS_BNT * 176 + 64);
  float* T1b   = carve((size_t)ROWS_BNT * 176 + 64);
  float* T2b   = carve((size_t)ROWS_BNT * 176 + 64);
  float* dataCL = carve((size_t)ROWS_BNT * 2 + 64);
  float* Z1    = carve((size_t)T_ * B_ * 128);
  _Float16* S2h = (_Float16*)carve((size_t)(T_ * B_) * NFEAT / 2);  // f16 spikes
  _Float16* Wtb = (_Float16*)carve((size_t)128 * NFEAT / 2);        // f16 weights
  float* statsb = carve(512);
  float* enorm  = carve(E_);
  float* W3p   = carve(256 * 48);
  float* B3p   = carve(48);
  int* csr_off = (int*)carve(128);
  int* csr_eid = (int*)carve(E_);
  if (ws_size < off * sizeof(float)) return;  // insufficient scratch

  k_graph<<<1, 256, 0, stream>>>(ei, enorm, csr_off, csr_eid);
  k_tr_data<<<(ROWS_BNT * 2 + 255) / 256, 256, 0, stream>>>(data, dataCL);

  auto gemm = [&](const float* X, int cit, int cio, int cin, const float* W,
                  const float* bias, float* Y, int cot, int coo, int cout,
                  bool accum) {
    int cinp = (cin + 31) & ~31;
    {
      int total = cout * cinp;
      k_prepw<<<(total + 255) / 256, 256, 0, stream>>>(W, Wtb, cin, cinp, cout);
    }
#define GEMM_LAUNCH(CO, NTW, GPM, AV)                                          \
    do {                                                                       \
      int waves = MT_BNT * (GPM);                                              \
      int blocks = (waves + 7) / 8;                                            \
      if (accum)                                                               \
        k_gemm_bnt<CO, NTW, true, AV><<<blocks, 256, 0, stream>>>(             \
            X, cit, cio, cin, cinp, Wtb, bias, Y, cot, coo);                   \
      else                                                                     \
        k_gemm_bnt<CO, NTW, false, AV><<<blocks, 256, 0, stream>>>(            \
            X, cit, cio, cin, cinp, Wtb, bias, Y, cot, coo);                   \
    } while (0)
    if (cout == 32)       GEMM_LAUNCH(32, 2, 1, true);
    else if (cout == 64)  GEMM_LAUNCH(64, 4, 1, false);  // conv0 (cit=2)
    else if (cout == 80)  GEMM_LAUNCH(80, 5, 1, true);
    else if (cout == 128) GEMM_LAUNCH(128, 4, 2, true);
#undef GEMM_LAUNCH
  };
  auto prop = [&](const float* X, int cit, int cio, int C, float* Y) {
    int total = ROWS_BNT * C;
    k_prop<<<(total + 255) / 256, 256, 0, stream>>>(X, cit, cio, C, Y, ei, enorm,
                                                    csr_off, csr_eid);
  };
  auto stats = [&](const float* X, int cit, int cio, int C, const float* g,
                   const float* b) {
    k_stats<<<C, 256, 0, stream>>>(X, cit, cio, g, b, statsb);
  };
  auto normlif = [&](const float* X, int cit, int cio, int C, float* Y, int cot,
                     int coo) {
    int total = B_ * N_ * C;
    k_normlif<<<(total + 255) / 256, 256, 0, stream>>>(X, cit, cio, C, Y, cot, coo,
                                                       statsb);
  };

  // conv0 (K=1, 2->64) -> raw into T2b, then norm0+lif -> BUF0 channels [0,64)
  gemm(dataCL, 2, 0, 2, P(75), P(74), T2b, 64, 0, 64, false);
  prop(dataCL, 2, 0, 2, T1b);
  gemm(T1b, 2, 0, 2, P(76), (const float*)nullptr, T2b, 64, 0, 64, true);
  stats(T2b, 64, 0, 64, P(84), P(83));
  normlif(T2b, 64, 0, 64, BUF0, 160, 0);

  const int nf0[2] = {64, 80};
  float* BUFs[2] = {BUF0, BUF1};
  const int CT[2] = {160, 176};
  for (int bi = 0; bi < 2; ++bi) {
    float* BUF = BUFs[bi];
    int ctot = CT[bi];
    for (int li = 0; li < 3; ++li) {
      int base = 2 + (bi * 3 + li) * 12;
      int cin = nf0[bi] + li * 32;
      // h = lif(ebnorm(x, norm1))
      stats(BUF, ctot, 0, cin, P(base + 9), P(base + 8));
      normlif(BUF, ctot, 0, cin, T0b, cin, 0);
      // hb = conv1(h)  (K=1: W0 h + W1 Ah + b)
      gemm(T0b, cin, 0, cin, P(base + 1), P(base + 0), T1b, 128, 0, 128, false);
      prop(T0b, cin, 0, cin, T2b);
      gemm(T2b, cin, 0, cin, P(base + 2), (const float*)nullptr, T1b, 128, 0, 128, true);
      // hs = lif(ebnorm(hb, norm2))
      stats(T1b, 128, 0, 128, P(base + 11), P(base + 10));
      normlif(T1b, 128, 0, 128, T0b, 128, 0);
      // conv2 (K=3) accumulated into BUF channel slice [cin, cin+32)
      gemm(T0b, 128, 0, 128, P(base + 4), P(base + 3), BUF, ctot, cin, 32, false);
      prop(T0b, 128, 0, 128, T1b);
      gemm(T1b, 128, 0, 128, P(base + 5), (const float*)nullptr, BUF, ctot, cin, 32, true);
      prop(T1b, 128, 0, 128, T2b);
      gemm(T2b, 128, 0, 128, P(base + 6), (const float*)nullptr, BUF, ctot, cin, 32, true);
      prop(T2b, 128, 0, 128, T1b);
      gemm(T1b, 128, 0, 128, P(base + 7), (const float*)nullptr, BUF, ctot, cin, 32, true);
    }
    if (bi == 0) {  // transition: 160 -> 80 into BUF1 channels [0,80)
      stats(BUF0, 160, 0, 160, P(91), P(90));
      normlif(BUF0, 160, 0, 160, T0b, 160, 0);
      gemm(T0b, 160, 0, 160, P(88), P(87), BUF1, 176, 0, 80, false);
      prop(T0b, 160, 0, 160, T1b);
      gemm(T1b, 160, 0, 160, P(89), (const float*)nullptr, BUF1, 176, 0, 80, true);
    }
  }

  // final normL + lif -> f16 spikes directly in fc1 layout [800, 17600]
  stats(BUF1, 176, 0, 176, P(86), P(85));
  {
    int total = B_ * N_ * 176;
    k_normlif_fc<<<(total + 255) / 256, 256, 0, stream>>>(BUF1, statsb, S2h);
  }

  // hoisted fc1 GEMM over all timesteps, then the sequential spiking head
  {
    int total = 128 * NFEAT;
    k_prepw<<<(total + 255) / 256, 256, 0, stream>>>(P(78), Wtb, NFEAT, NFEAT, 128);
  }
  {
    int waves = T_ * 2;
    k_gemm_tb<<<(waves + 7) / 8, 256, 0, stream>>>(S2h, Wtb, P(77), Z1);
  }
  k_pad_fc3<<<(256 * 48 + 255) / 256, 256, 0, stream>>>(P(82), P(81), W3p, B3p);
  k_fc_scan<<<1, 256, 0, stream>>>(Z1, P(80), P(79), W3p, B3p, (float*)d_out);
}
